// EncodeProcessDecode_12309376270350
// MI455X (gfx1250) — matrix operand order repaired and checked by reading, not compiled, not run on hardware
//
#include <hip/hip_runtime.h>
#include <hip/hip_bf16.h>

#define N_NODES 50000
#define N_EDGES 800000

typedef __attribute__((ext_vector_type(16))) _Float16 v16h;
typedef __attribute__((ext_vector_type(8)))  float    v8f;

static __device__ inline v8f wmma16(v16h a, v16h b, v8f c) {
  return __builtin_amdgcn_wmma_f32_16x16x32_f16(false, a, false, b, (short)0, c, false, false);
}

// ---- A fragment (16x32 f16, MxK) from a per-lane f16 row pointer (row m = lane&15).
// Per lane the 16 halves form two contiguous 8-half (16B) runs -> 2x b128 loads.
static __device__ inline v16h a_frag_rowptr_h(const _Float16* __restrict__ rp, int k0) {
  const int lane = threadIdx.x & 31;
  const int kh = (lane & 16) ? 8 : 0;
  v16h a;
#pragma unroll
  for (int r = 0; r < 8; ++r) {
    int k = k0 + ((r & 4) ? 16 : 0) + kh + ((r & 3) << 1);
    a[2 * r]     = rp[k];
    a[2 * r + 1] = rp[k + 1];
  }
  return a;
}

// ---- Same, from an f32 row (used for the f32 mean-aggregate h_N).
static __device__ inline v16h a_frag_rowptr_f(const float* __restrict__ rp, int k0) {
  const int lane = threadIdx.x & 31;
  const int kh = (lane & 16) ? 8 : 0;
  v16h a;
#pragma unroll
  for (int r = 0; r < 8; ++r) {
    int k = k0 + ((r & 4) ? 16 : 0) + kh + ((r & 3) << 1);
    a[2 * r]     = (_Float16)rp[k];
    a[2 * r + 1] = (_Float16)rp[k + 1];
  }
  return a;
}

// ---- A fragment from a 16-row f32 LDS tile (leading dim ld), K masked to kmax.
static __device__ inline v16h a_frag_tile(const float* t, int ld, int k0, int kmax) {
  const int lane = threadIdx.x & 31;
  const int m = lane & 15;
  const int kh = (lane & 16) ? 8 : 0;
  const float* rp = t + m * ld;
  v16h a;
#pragma unroll
  for (int r = 0; r < 8; ++r) {
    int k = k0 + ((r & 4) ? 16 : 0) + kh + ((r & 3) << 1);
    a[2 * r]     = (_Float16)((k     < kmax) ? rp[k]     : 0.0f);
    a[2 * r + 1] = (_Float16)((k + 1 < kmax) ? rp[k + 1] : 0.0f);
  }
  return a;
}

// ---- Pre-packed B fragment loaders (32B-contiguous per lane).
static __device__ inline v16h load_frag(const _Float16* __restrict__ base, int f) {
  const int lane = threadIdx.x & 31;
  return *reinterpret_cast<const v16h*>(base + (size_t)f * 512 + (size_t)lane * 16);
}

// ---- C/D layout: lane = N (lane&15), VGPR r -> M = r + (lane>=16 ? 8 : 0).
static __device__ inline void store_tile_relu(float* t, int ld, v8f c, float bb) {
  const int lane = threadIdx.x & 31;
  const int n = lane & 15;
  const int mh = (lane & 16) ? 8 : 0;
#pragma unroll
  for (int r = 0; r < 8; ++r) {
    float v = c[r] + bb;
    t[(mh + r) * ld + n] = v > 0.0f ? v : 0.0f;
  }
}

static __device__ inline void store_out_sigmoid_h(_Float16* ob, int ldo, v8f c, float bb) {
  const int lane = threadIdx.x & 31;
  const int n = lane & 15;
  const int mh = (lane & 16) ? 8 : 0;
#pragma unroll
  for (int r = 0; r < 8; ++r) {
    float v = c[r] + bb;
    ob[(size_t)(mh + r) * ldo + n] = (_Float16)(1.0f / (1.0f + __expf(-v)));
  }
}

// ---- latent-16 tail (L2 + L3), weight frags in registers.
static __device__ inline void mlp16_tail_r(v16h f2, float b2v,
                                           v16h f30, v16h f31, v16h f32v, v16h f33,
                                           float b3v0, float b3v1, float b3v2, float b3v3,
                                           float* hbuf, float* hbuf2, _Float16* outbase) {
  v8f a2 = {};
  a2 = wmma16(a_frag_tile(hbuf, 17, 0, 16), f2, a2);
  store_tile_relu(hbuf2, 17, a2, b2v);
  __syncthreads();
  v16h a3 = a_frag_tile(hbuf2, 17, 0, 16);
  v8f o;
  o = v8f{}; o = wmma16(a3, f30, o); store_out_sigmoid_h(outbase +  0, 64, o, b3v0);
  o = v8f{}; o = wmma16(a3, f31, o); store_out_sigmoid_h(outbase + 16, 64, o, b3v1);
  o = v8f{}; o = wmma16(a3, f32v, o); store_out_sigmoid_h(outbase + 32, 64, o, b3v2);
  o = v8f{}; o = wmma16(a3, f33, o); store_out_sigmoid_h(outbase + 48, 64, o, b3v3);
}

// ================= weight fragment packing =================
__global__ void k_pack(const float* __restrict__ W, int ldn, int kmax, int nfrag_k,
                       _Float16* __restrict__ out) {
  const int lane = threadIdx.x & 31;
  const int f = blockIdx.x;
  const int kstep = f % nfrag_k;
  const int ntile = f / nfrag_k;
  const int n = ntile * 16 + (lane & 15);
  const int kg = (lane & 16) ? 8 : 0;
  _Float16* o = out + (size_t)f * 512 + (size_t)lane * 16;
  for (int j = 0; j < 16; ++j) {
    int k = kstep * 32 + ((j & 8) ? 16 : 0) + kg + (j & 7);
    float v = (k < kmax) ? W[(size_t)k * ldn + n] : 0.0f;
    o[j] = (_Float16)v;
  }
}

// ================= encoders =================
__global__ __launch_bounds__(160) void k_encode_nodes(
    const float* __restrict__ C, const float* __restrict__ F,
    const _Float16* __restrict__ pw, const float* b1, const float* b2, const float* b3,
    _Float16* __restrict__ hn, int tpw) {
  __shared__ float xt[5][16 * 4];
  __shared__ float h1[5][16 * 17];
  __shared__ float h2[5][16 * 17];
  const int w = threadIdx.x >> 5, lane = threadIdx.x & 31;
  const int n = lane & 15;
  v16h f1 = load_frag(pw, 0), f2 = load_frag(pw, 1);
  v16h f30 = load_frag(pw, 2), f31 = load_frag(pw, 3), f32v = load_frag(pw, 4), f33 = load_frag(pw, 5);
  const float b1v = b1[n], b2v = b2[n];
  const float b3v0 = b3[n], b3v1 = b3[16 + n], b3v2 = b3[32 + n], b3v3 = b3[48 + n];
  const int tbase = (blockIdx.x * 5 + w) * tpw;
#pragma unroll 1
  for (int t = 0; t < tpw; ++t) {
    const size_t row0 = (size_t)(tbase + t) * 16;
    if (lane < 16) {
      xt[w][lane * 4 + 0] = C[row0 + lane];
      xt[w][lane * 4 + 1] = F[row0 + lane];
    }
    __syncthreads();
    v8f acc = {};
    acc = wmma16(a_frag_tile(xt[w], 4, 0, 2), f1, acc);
    store_tile_relu(h1[w], 17, acc, b1v);
    __syncthreads();
    mlp16_tail_r(f2, b2v, f30, f31, f32v, f33, b3v0, b3v1, b3v2, b3v3,
                 h1[w], h2[w], hn + row0 * 64);
  }
}

__global__ __launch_bounds__(256) void k_encode_edges(
    const float* __restrict__ A, const float* __restrict__ S1, const float* __restrict__ S0,
    const _Float16* __restrict__ pw, const float* b1, const float* b2, const float* b3,
    _Float16* __restrict__ he, int tpw) {
  __shared__ float xt[8][16 * 4];
  __shared__ float h1[8][16 * 17];
  __shared__ float h2[8][16 * 17];
  const int w = threadIdx.x >> 5, lane = threadIdx.x & 31;
  const int n = lane & 15;
  v16h f1 = load_frag(pw, 0), f2 = load_frag(pw, 1);
  v16h f30 = load_frag(pw, 2), f31 = load_frag(pw, 3), f32v = load_frag(pw, 4), f33 = load_frag(pw, 5);
  const float b1v = b1[n], b2v = b2[n];
  const float b3v0 = b3[n], b3v1 = b3[16 + n], b3v2 = b3[32 + n], b3v3 = b3[48 + n];
  const int tbase = (blockIdx.x * 8 + w) * tpw;
#pragma unroll 1
  for (int t = 0; t < tpw; ++t) {
    const size_t row0 = (size_t)(tbase + t) * 16;
    if (lane < 16) {
      xt[w][lane * 4 + 0] = A[row0 + lane];
      xt[w][lane * 4 + 1] = S1[row0 + lane];
      xt[w][lane * 4 + 2] = S0[row0 + lane];
    }
    __syncthreads();
    v8f acc = {};
    acc = wmma16(a_frag_tile(xt[w], 4, 0, 3), f1, acc);
    store_tile_relu(h1[w], 17, acc, b1v);
    __syncthreads();
    mlp16_tail_r(f2, b2v, f30, f31, f32v, f33, b3v0, b3v1, b3v2, b3v3,
                 h1[w], h2[w], he + row0 * 64);
  }
}

// ================= GN block stages =================
__global__ __launch_bounds__(256) void k_edge_update(
    const _Float16* __restrict__ he_in, const _Float16* __restrict__ hn,
    const int* __restrict__ src, const int* __restrict__ dst,
    const _Float16* __restrict__ pw, const float* b1, const float* b2, const float* b3,
    _Float16* __restrict__ he_out, int tpw) {
  __shared__ float h1[8][16 * 17];
  __shared__ float h2[8][16 * 17];
  const int w = threadIdx.x >> 5, lane = threadIdx.x & 31;
  const int n = lane & 15;
  v16h f10 = load_frag(pw, 0), f11 = load_frag(pw, 1), f12 = load_frag(pw, 2);
  v16h f13 = load_frag(pw, 3), f14 = load_frag(pw, 4), f15 = load_frag(pw, 5);
  v16h f2 = load_frag(pw, 6);
  v16h f30 = load_frag(pw, 7), f31 = load_frag(pw, 8), f32v = load_frag(pw, 9), f33 = load_frag(pw, 10);
  const float b1v = b1[n], b2v = b2[n];
  const float b3v0 = b3[n], b3v1 = b3[16 + n], b3v2 = b3[32 + n], b3v3 = b3[48 + n];
  const int tbase = (blockIdx.x * 8 + w) * tpw;
#pragma unroll 1
  for (int t = 0; t < tpw; ++t) {
    const size_t e0 = (size_t)(tbase + t) * 16;
    const size_t e = e0 + n;
    const _Float16* pe = he_in + e * 64;
    const _Float16* ps = hn + (size_t)src[e] * 64;
    const _Float16* pd = hn + (size_t)dst[e] * 64;
    v8f acc = {};
    acc = wmma16(a_frag_rowptr_h(pe, 0),  f10, acc);
    acc = wmma16(a_frag_rowptr_h(pe, 32), f11, acc);
    acc = wmma16(a_frag_rowptr_h(ps, 0),  f12, acc);
    acc = wmma16(a_frag_rowptr_h(ps, 32), f13, acc);
    acc = wmma16(a_frag_rowptr_h(pd, 0),  f14, acc);
    acc = wmma16(a_frag_rowptr_h(pd, 32), f15, acc);
    store_tile_relu(h1[w], 17, acc, b1v);
    __syncthreads();
    mlp16_tail_r(f2, b2v, f30, f31, f32v, f33, b3v0, b3v1, b3v2, b3v3,
                 h1[w], h2[w], he_out + e0 * 64);
  }
}

__global__ void k_clear(float* __restrict__ sum, float* __restrict__ cnt) {
  size_t i = (size_t)blockIdx.x * 256 + threadIdx.x;  // i < N_NODES*64
  sum[i] = 0.0f;
  if (i < N_NODES) cnt[i] = 0.0f;
}

__global__ void k_scatter(const _Float16* __restrict__ he, const int* __restrict__ dst,
                          float* __restrict__ sum, float* __restrict__ cnt) {
  size_t i = (size_t)blockIdx.x * 256 + threadIdx.x;  // i < N_EDGES*64
  int e = (int)(i >> 6);
  int f = (int)(i & 63);
  int d = dst[e];
  atomicAdd(&sum[(size_t)d * 64 + f], (float)he[i]);
  if (f == 0) atomicAdd(&cnt[d], 1.0f);
}

__global__ void k_mean(const float* __restrict__ sum, const float* __restrict__ cnt,
                       float* __restrict__ hN) {
  size_t i = (size_t)blockIdx.x * 256 + threadIdx.x;  // i < N_NODES*64
  hN[i] = sum[i] / fmaxf(cnt[i >> 6], 1.0f);
}

__global__ __launch_bounds__(160) void k_node_update(
    const _Float16* __restrict__ hn_in, const float* __restrict__ hN,
    const _Float16* __restrict__ pw, const float* b1, const float* b2, const float* b3,
    _Float16* __restrict__ hn_out, int tpw) {
  __shared__ float h1[5][16 * 17];
  __shared__ float h2[5][16 * 17];
  const int w = threadIdx.x >> 5, lane = threadIdx.x & 31;
  const int n = lane & 15;
  v16h f10 = load_frag(pw, 0), f11 = load_frag(pw, 1), f12 = load_frag(pw, 2), f13 = load_frag(pw, 3);
  v16h f2 = load_frag(pw, 4);
  v16h f30 = load_frag(pw, 5), f31 = load_frag(pw, 6), f32v = load_frag(pw, 7), f33 = load_frag(pw, 8);
  const float b1v = b1[n], b2v = b2[n];
  const float b3v0 = b3[n], b3v1 = b3[16 + n], b3v2 = b3[32 + n], b3v3 = b3[48 + n];
  const int tbase = (blockIdx.x * 5 + w) * tpw;
#pragma unroll 1
  for (int t = 0; t < tpw; ++t) {
    const size_t r0 = (size_t)(tbase + t) * 16;
    const size_t row = r0 + n;
    const _Float16* pn = hn_in + row * 64;
    const float* pa = hN + row * 64;
    v8f acc = {};
    acc = wmma16(a_frag_rowptr_h(pn, 0),  f10, acc);
    acc = wmma16(a_frag_rowptr_h(pn, 32), f11, acc);
    acc = wmma16(a_frag_rowptr_f(pa, 0),  f12, acc);
    acc = wmma16(a_frag_rowptr_f(pa, 32), f13, acc);
    store_tile_relu(h1[w], 17, acc, b1v);
    __syncthreads();
    mlp16_tail_r(f2, b2v, f30, f31, f32v, f33, b3v0, b3v1, b3v2, b3v3,
                 h1[w], h2[w], hn_out + r0 * 64);
  }
}

// ================= decoder (192 -> 64 -> 64 -> 1) =================
// Weight frags staged in LDS once per block; L2+L3 fused via shfl reduction.
__global__ __launch_bounds__(128) void k_decode(
    const _Float16* __restrict__ he, const _Float16* __restrict__ hn,
    const int* __restrict__ src, const int* __restrict__ dst,
    const _Float16* __restrict__ pw,  // 24 L1 frags then 8 L2 frags
    const float* b1, const float* b2,
    const float* w3, const float* b3, float* __restrict__ out, int tpw) {
  __shared__ __align__(32) _Float16 wf[32 * 512];  // 32 KB of packed frags
  __shared__ float h1[4][16 * 65];                 // 16.6 KB
  const int w = threadIdx.x >> 5, lane = threadIdx.x & 31;
  const int n = lane & 15;
  {
    const uint4* gs = (const uint4*)pw;
    uint4* gd = (uint4*)wf;
    for (int i = threadIdx.x; i < 2048; i += 128) gd[i] = gs[i];
  }
  __syncthreads();
  float b1v[4], b2v[4], w3v[4];
#pragma unroll
  for (int nt = 0; nt < 4; ++nt) {
    b1v[nt] = b1[nt * 16 + n];
    b2v[nt] = b2[nt * 16 + n];
    w3v[nt] = w3[nt * 16 + n];
  }
  const float bb3 = b3[0];
  const int tbase = (blockIdx.x * 4 + w) * tpw;
#pragma unroll 1
  for (int t = 0; t < tpw; ++t) {
    const size_t e0 = (size_t)(tbase + t) * 16;
    const size_t e = e0 + n;
    const _Float16* pe = he + e * 64;
    const _Float16* ps = hn + (size_t)src[e] * 64;
    const _Float16* pd = hn + (size_t)dst[e] * 64;
    v16h af[6];
    af[0] = a_frag_rowptr_h(pe, 0);  af[1] = a_frag_rowptr_h(pe, 32);
    af[2] = a_frag_rowptr_h(ps, 0);  af[3] = a_frag_rowptr_h(ps, 32);
    af[4] = a_frag_rowptr_h(pd, 0);  af[5] = a_frag_rowptr_h(pd, 32);
#pragma unroll
    for (int nt = 0; nt < 4; ++nt) {
      v8f acc = {};
#pragma unroll
      for (int j = 0; j < 6; ++j)
        acc = wmma16(af[j], load_frag(wf, nt * 6 + j), acc);
      store_tile_relu(&h1[w][nt * 16], 65, acc, b1v[nt]);
    }
    __syncthreads();
    v16h a0 = a_frag_tile(h1[w], 65, 0, 64);
    v16h a1 = a_frag_tile(h1[w], 65, 32, 64);
    // fused L2 (WMMA) + L3 (relu * w3, per-row partials)
    float p[8];
#pragma unroll
    for (int r = 0; r < 8; ++r) p[r] = 0.0f;
#pragma unroll
    for (int nt = 0; nt < 4; ++nt) {
      v8f acc = {};
      acc = wmma16(a0, load_frag(wf, 24 + nt * 2 + 0), acc);
      acc = wmma16(a1, load_frag(wf, 24 + nt * 2 + 1), acc);
#pragma unroll
      for (int r = 0; r < 8; ++r) {
        float v = acc[r] + b2v[nt];
        p[r] += (v > 0.0f ? v : 0.0f) * w3v[nt];
      }
    }
    // reduce over the 16 lanes of each half-wave (columns of each row group)
#pragma unroll
    for (int off = 1; off < 16; off <<= 1) {
#pragma unroll
      for (int r = 0; r < 8; ++r) p[r] += __shfl_xor(p[r], off, 32);
    }
    if (n == 0) {  // lanes 0 and 16: rows 0..7 and 8..15
      const int mh = (lane & 16) ? 8 : 0;
#pragma unroll
      for (int r = 0; r < 8; ++r)
        out[e0 + mh + r] = 1.0f / (1.0f + __expf(-(p[r] + bb3)));
    }
  }
}

extern "C" void kernel_launch(void* const* d_in, const int* in_sizes, int n_in,
                              void* d_out, int out_size, void* d_ws, size_t ws_size,
                              hipStream_t stream) {
  (void)in_sizes; (void)n_in; (void)out_size; (void)ws_size;
  const float* C  = (const float*)d_in[0];
  const float* F  = (const float*)d_in[1];
  const float* A  = (const float*)d_in[2];
  const float* S1 = (const float*)d_in[3];
  const float* S0 = (const float*)d_in[4];
  const int*  src = (const int*)d_in[5];
  const int*  dst = (const int*)d_in[6];
  // param groups: 0=enc_n 1=enc_e 2=c1e 3=c1n 4=c3e 5=c3n 6=dec ; leaves [w1,b1,w2,b2,w3,b3]
  auto P = [&](int g, int j) { return (const float*)d_in[7 + g * 6 + j]; };

  _Float16* hn = (_Float16*)d_ws;                            // N*64 f16
  _Float16* he = hn + (size_t)N_NODES * 64;                  // E*64 f16
  float* sum = (float*)(he + (size_t)N_EDGES * 64);          // N*64 f32
  float* cnt = sum + (size_t)N_NODES * 64;                   // N f32
  float* hN  = cnt + N_NODES;                                // N*64 f32
  _Float16* arena = (_Float16*)(hN + (size_t)N_NODES * 64);  // packed weight frags

  const int nk1[6]      = {1, 1, 6, 4, 6, 4};
  const int kin[6]      = {2, 3, 192, 128, 192, 128};
  const int baseFrag[7] = {0, 6, 12, 23, 32, 41, 50};

  // ---- pack weights into WMMA B-fragment layout ----
  for (int g = 0; g < 6; ++g) {
    _Float16* base = arena + (size_t)baseFrag[g] * 512;
    k_pack<<<nk1[g], 32, 0, stream>>>(P(g, 0), 16, kin[g], nk1[g], base);
    k_pack<<<1, 32, 0, stream>>>(P(g, 2), 16, 16, 1, base + (size_t)nk1[g] * 512);
    k_pack<<<4, 32, 0, stream>>>(P(g, 4), 64, 16, 1, base + (size_t)(nk1[g] + 1) * 512);
  }
  _Float16* dbase = arena + (size_t)baseFrag[6] * 512;
  k_pack<<<24, 32, 0, stream>>>(P(6, 0), 64, 192, 6, dbase);
  k_pack<<<8, 32, 0, stream>>>(P(6, 2), 64, 64, 2, dbase + (size_t)24 * 512);

  const int NT_N = N_NODES / 16;  // 3125
  const int NT_E = N_EDGES / 16;  // 50000
  const int TPW_N = 5;            // 3125 = 125 blocks * 5 waves * 5 tiles
  const int TPW_E = 10;           // 50000 = 625 blocks * 8 waves * 10 tiles
  const int TPW_D = 10;           // 50000 = 1250 blocks * 4 waves * 10 tiles

  k_encode_nodes<<<NT_N / (5 * TPW_N), 160, 0, stream>>>(C, F,
      arena + (size_t)baseFrag[0] * 512, P(0, 1), P(0, 3), P(0, 5), hn, TPW_N);
  k_encode_edges<<<NT_E / (8 * TPW_E), 256, 0, stream>>>(A, S1, S0,
      arena + (size_t)baseFrag[1] * 512, P(1, 1), P(1, 3), P(1, 5), he, TPW_E);

  for (int blk = 0; blk < 2; ++blk) {
    const int ge = (blk == 0) ? 2 : 4;
    const int gn = (blk == 0) ? 3 : 5;
    k_edge_update<<<NT_E / (8 * TPW_E), 256, 0, stream>>>(he, hn, src, dst,
        arena + (size_t)baseFrag[ge] * 512, P(ge, 1), P(ge, 3), P(ge, 5), he, TPW_E);
    k_clear<<<(N_NODES * 64) / 256, 256, 0, stream>>>(sum, cnt);
    k_scatter<<<(N_EDGES * 64) / 256, 256, 0, stream>>>(he, dst, sum, cnt);
    k_mean<<<(N_NODES * 64) / 256, 256, 0, stream>>>(sum, cnt, hN);
    k_node_update<<<NT_N / (5 * TPW_N), 160, 0, stream>>>(hn, hN,
        arena + (size_t)baseFrag[gn] * 512, P(gn, 1), P(gn, 3), P(gn, 5), hn, TPW_N);
  }

  k_decode<<<NT_E / (4 * TPW_D), 128, 0, stream>>>(he, hn, src, dst,
      dbase, P(6, 1), P(6, 3), P(6, 4), P(6, 5), (float*)d_out, TPW_D);
}
